// MHARoPE_51393578664494
// MI455X (gfx1250) — compile-verified, matrix operand order checked by reading, not run on hardware
//
#include <hip/hip_runtime.h>
#include <hip/hip_bf16.h>
#include <cstdint>

// ---------------------------------------------------------------------------
// MHA + RoPE for MI455X (gfx1250), bf16 WMMA path, fp32 accumulate.
// B=4, S=2048, E=1024, H=16, D=64.
// GEMMs: double-buffered GLOBAL_LOAD_ASYNC_TO_LDS_B128 (ASYNCcnt) staging,
// manually unrolled ping-pong, one barrier per K-tile.
// ---------------------------------------------------------------------------

typedef __bf16 v16bf __attribute__((ext_vector_type(16)));
typedef float  v8f   __attribute__((ext_vector_type(8)));

#define B_   4
#define S_   2048
#define E_   1024
#define H_   16
#define D_   64
#define KD_  1024           // GEMM K dimension
#define M_   (B_ * S_)      // 8192 GEMM rows
#define NEG_INF_ (-1e30f)
#define LN1E4_OVER32 0.28782313662425572f   // ln(10000)/32

__device__ __forceinline__ unsigned short f2bf(float f) {
  unsigned u = __builtin_bit_cast(unsigned, f);
  u += 0x7FFFu + ((u >> 16) & 1u);          // round-to-nearest-even
  return (unsigned short)(u >> 16);
}

__device__ __forceinline__ v16bf ldfrag(const unsigned short* p) {
  return *reinterpret_cast<const v16bf*>(p);  // 32B, callers guarantee alignment
}

__device__ __forceinline__ v8f wmma_bf16(v16bf a, v16bf b, v8f c) {
  // D = A(16x32 bf16) * B(32x16 bf16) + C(16x16 f32)
  return __builtin_amdgcn_wmma_f32_16x16x32_bf16(
      /*neg_a=*/false, a, /*neg_b=*/false, b,
      /*c_mod=*/(short)0, c, /*reuse_a=*/false, /*reuse_b=*/false);
}

// Async global->LDS copy, 16B per lane, tracked by ASYNCcnt.
// lds_off: byte offset within the workgroup LDS allocation (low 32 bits of a
// generic pointer to __shared__, per the aperture rules).
__device__ __forceinline__ void async_ld_b128(unsigned lds_off, const void* g) {
  asm volatile("global_load_async_to_lds_b128 %0, %1, off"
               :: "v"(lds_off), "v"(g) : "memory");
}
#define WAIT_ASYNC(n) asm volatile("s_wait_asynccnt " #n ::: "memory")

__device__ __forceinline__ float redmax16(float v) {
  v = fmaxf(v, __shfl_xor(v, 1, 16));
  v = fmaxf(v, __shfl_xor(v, 2, 16));
  v = fmaxf(v, __shfl_xor(v, 4, 16));
  v = fmaxf(v, __shfl_xor(v, 8, 16));
  return v;
}
__device__ __forceinline__ float redsum16(float v) {
  v += __shfl_xor(v, 1, 16);
  v += __shfl_xor(v, 2, 16);
  v += __shfl_xor(v, 4, 16);
  v += __shfl_xor(v, 8, 16);
  return v;
}

// ---------------------------------------------------------------------------
// fp32 -> bf16 bulk convert (n is a multiple of 4)
// ---------------------------------------------------------------------------
__global__ __launch_bounds__(256)
void cvt_bf16_k(const float* __restrict__ in, unsigned short* __restrict__ out,
                int n4) {
  int i = blockIdx.x * 256 + threadIdx.x;
  if (i < n4) {
    float4 v = reinterpret_cast<const float4*>(in)[i];
    ushort4 r;
    r.x = f2bf(v.x); r.y = f2bf(v.y); r.z = f2bf(v.z); r.w = f2bf(v.w);
    reinterpret_cast<ushort4*>(out)[i] = r;
  }
}

// ---------------------------------------------------------------------------
// Tiled WMMA GEMM:  out[m,n] = sum_k A[m,k] * W[n,k]   (A row-major, W row-major)
// Block tile 128(M) x 64(N), 8 waves; double-buffered async LDS staging.
// mode 0: Q  -> RoPE epilogue, bf16 [B,H,S,D]
// mode 1: K  -> RoPE epilogue, bf16 [B,H,S,D]
// mode 2: V  -> bf16 transposed [B,H,D,S]
// mode 3: O  -> + bias, fp32 row-major [B*S, E]
// ---------------------------------------------------------------------------
__global__ __launch_bounds__(256)
void gemm_wmma_k(const unsigned short* __restrict__ A,
                 const unsigned short* __restrict__ W,
                 unsigned short* __restrict__ outBF,
                 float* __restrict__ outF32,
                 const float* __restrict__ dt,
                 const float* __restrict__ bo,
                 int mode) {
  // double-buffered tiles, row stride 48 elems (96B) for bank staggering
  __shared__ __align__(32) unsigned short xs[2][128 * 48];  // 2 x 12 KB
  __shared__ __align__(32) unsigned short wsh[2][64 * 48];  // 2 x  6 KB

  const int tid  = threadIdx.x;
  const int wave = tid >> 5;
  const int lane = tid & 31;
  const int ln   = lane & 15;    // C-layout column within 16-tile
  const int hi   = lane >> 4;    // half-wave: rows r vs r+8
  const int bm   = blockIdx.y * 128;
  const int bn   = blockIdx.x * 64;

  // per-lane async-copy geometry: lane covers (row group, 16B segment)
  const int lrow = lane >> 2;    // 0..7
  const int lseg = lane & 3;     // 0..3  (seg*8 bf16 = 16B)

  v8f acc0 = v8f{0.f,0.f,0.f,0.f,0.f,0.f,0.f,0.f};
  v8f acc1 = acc0, acc2 = acc0, acc3 = acc0;

  // Each wave issues 3 async b128 copies per tile:
  //   A rows [wave*16, wave*16+16): 2 instrs (8 rows x 4 segs each)
  //   W rows [wave*8,  wave*8+8):   1 instr
  auto issue_tile = [&](int buf, int k0) {
    const int ar0 = wave * 16 + lrow;
    async_ld_b128((unsigned)(uintptr_t)&xs[buf][ar0 * 48 + lseg * 8],
                  A + (size_t)(bm + ar0) * KD_ + k0 + lseg * 8);
    const int ar1 = ar0 + 8;
    async_ld_b128((unsigned)(uintptr_t)&xs[buf][ar1 * 48 + lseg * 8],
                  A + (size_t)(bm + ar1) * KD_ + k0 + lseg * 8);
    const int wr = wave * 8 + lrow;
    async_ld_b128((unsigned)(uintptr_t)&wsh[buf][wr * 48 + lseg * 8],
                  W + (size_t)(bn + wr) * KD_ + k0 + lseg * 8);
  };

  auto compute = [&](int buf) {
    v16bf a = ldfrag(&xs[buf][(wave * 16 + ln) * 48 + hi * 16]);
    acc0 = wmma_bf16(a, ldfrag(&wsh[buf][(0 * 16 + ln) * 48 + hi * 16]), acc0);
    acc1 = wmma_bf16(a, ldfrag(&wsh[buf][(1 * 16 + ln) * 48 + hi * 16]), acc1);
    acc2 = wmma_bf16(a, ldfrag(&wsh[buf][(2 * 16 + ln) * 48 + hi * 16]), acc2);
    acc3 = wmma_bf16(a, ldfrag(&wsh[buf][(3 * 16 + ln) * 48 + hi * 16]), acc3);
  };

  // Ping-pong pipeline, manually unrolled by 2 (nIter is even).
  // Invariant at each compute(buf): buf's fills are complete (waited+barriered)
  // and the opposite buffer's previous readers are past the last barrier
  // (a wave can only pass it after s_wait_dscnt 0 preceding its WMMAs).
  const int nIter = KD_ / 32;   // 32
  issue_tile(0, 0);
  WAIT_ASYNC(0x0);
  __syncthreads();
#pragma unroll 1
  for (int it = 0; it < nIter; it += 2) {
    issue_tile(1, (it + 1) * 32);       // overlap buf1 fill with buf0 compute
    compute(0);
    WAIT_ASYNC(0x0);
    __syncthreads();
    if (it + 2 < nIter) issue_tile(0, (it + 2) * 32);
    compute(1);
    if (it + 2 < nIter) {
      WAIT_ASYNC(0x0);
      __syncthreads();
    }
  }

  // ---- epilogue ----
  const v8f accs[4] = {acc0, acc1, acc2, acc3};
#pragma unroll
  for (int t = 0; t < 4; ++t) {
    const int col = bn + t * 16 + ln;       // output feature
#pragma unroll
    for (int r = 0; r < 8; ++r) {
      const int mrow = bm + wave * 16 + r + hi * 8;   // global row (b*S+s)
      float v = accs[t][r];
      if (mode <= 1) {
        // RoPE: pairs are (even col, odd col) = adjacent lanes
        const int b = mrow >> 11, s = mrow & (S_ - 1);
        const int d = col & (D_ - 1), h = col >> 6;
        const int p = d >> 1;
        float inv_freq = __expf(-(float)p * LN1E4_OVER32);
        float ang = dt[(b << 11) + s] * inv_freq;
        float sn, cs;
        __sincosf(ang, &sn, &cs);
        float other = __shfl_xor(v, 1, 32);
        float res = (d & 1) ? (other * sn + v * cs)   // r1 = x0*sin + x1*cos
                            : (v * cs - other * sn);  // r0 = x0*cos - x1*sin
        outBF[((size_t)(b * H_ + h) * S_ + s) * D_ + d] = f2bf(res);
      } else if (mode == 2) {
        const int b = mrow >> 11, s = mrow & (S_ - 1);
        const int d = col & (D_ - 1), h = col >> 6;
        outBF[((size_t)(b * H_ + h) * D_ + d) * S_ + s] = f2bf(v);  // V^T
      } else {
        outF32[(size_t)mrow * E_ + col] = v + bo[col];
      }
    }
  }
}

// ---------------------------------------------------------------------------
// Flash attention: grid (S/64, H, B), 128 threads = 4 waves, 16 query rows/wave.
// Q,K: bf16 [B,H,S,D]; VT: bf16 [B,H,D,S]; OUT: bf16 [B,S,E] (= [B,S,H*D]).
// ---------------------------------------------------------------------------
__global__ __launch_bounds__(128)
void attn_wmma_k(const unsigned short* __restrict__ Q,
                 const unsigned short* __restrict__ K,
                 const unsigned short* __restrict__ VT,
                 const unsigned char* __restrict__ mask,
                 unsigned short* __restrict__ OUT) {
  __shared__ __align__(32) unsigned short pbuf[4][16 * 32];  // P tile per wave

  const int tid  = threadIdx.x;
  const int wave = tid >> 5;
  const int lane = tid & 31;
  const int ln   = lane & 15;
  const int hi   = lane >> 4;
  const int b = blockIdx.z, h = blockIdx.y;
  const int rowbase = blockIdx.x * 64 + wave * 16;

  const unsigned short* qb = Q  + ((size_t)(b * H_ + h) * S_ + rowbase) * D_;
  const unsigned short* kp = K  + ((size_t)(b * H_ + h) * S_) * D_;
  const unsigned short* vp = VT + ((size_t)(b * H_ + h) * D_) * S_;
  const unsigned char*  mp = mask + (size_t)b * S_;
  unsigned short* pw = pbuf[wave];

  // Q A-fragments (16x64 = two 16x32 K-slices)
  v16bf aq0 = ldfrag(qb + (size_t)ln * D_ + hi * 16);
  v16bf aq1 = ldfrag(qb + (size_t)ln * D_ + 32 + hi * 16);

  v8f o0 = v8f{0.f,0.f,0.f,0.f,0.f,0.f,0.f,0.f};
  v8f o1 = o0, o2 = o0, o3 = o0;
  float mstat[8], lstat[8];
#pragma unroll
  for (int r = 0; r < 8; ++r) { mstat[r] = -3.0e38f; lstat[r] = 0.f; }

  const float scale = 0.125f;  // 1/sqrt(64)

  for (int kb = 0; kb < S_; kb += 32) {
    // ---- scores: S(16x32) = Q(16x64) * K^T, two 16x16 column tiles ----
    v8f s0 = v8f{0.f,0.f,0.f,0.f,0.f,0.f,0.f,0.f};
    v8f s1 = s0;
    {
      v16bf bk;
      bk = ldfrag(kp + (size_t)(kb + ln) * D_ + hi * 16);        s0 = wmma_bf16(aq0, bk, s0);
      bk = ldfrag(kp + (size_t)(kb + ln) * D_ + 32 + hi * 16);   s0 = wmma_bf16(aq1, bk, s0);
      bk = ldfrag(kp + (size_t)(kb + 16 + ln) * D_ + hi * 16);      s1 = wmma_bf16(aq0, bk, s1);
      bk = ldfrag(kp + (size_t)(kb + 16 + ln) * D_ + 32 + hi * 16); s1 = wmma_bf16(aq1, bk, s1);
    }
    const bool k0m = mp[kb + ln] != 0;
    const bool k1m = mp[kb + 16 + ln] != 0;

    // ---- online softmax per row (rows live across the 16-lane half-wave) ----
#pragma unroll
    for (int r = 0; r < 8; ++r) {
      float v0 = k0m ? NEG_INF_ : s0[r] * scale;
      float v1 = k1m ? NEG_INF_ : s1[r] * scale;
      float mx = redmax16(fmaxf(v0, v1));
      float mnew = fmaxf(mstat[r], mx);
      float alpha = __expf(mstat[r] - mnew);
      float p0 = __expf(v0 - mnew);
      float p1 = __expf(v1 - mnew);
      lstat[r] = lstat[r] * alpha + redsum16(p0 + p1);
      mstat[r] = mnew;
      o0[r] *= alpha; o1[r] *= alpha; o2[r] *= alpha; o3[r] *= alpha;
      const int prow = r + hi * 8;
      pw[prow * 32 + ln]      = f2bf(p0);
      pw[prow * 32 + 16 + ln] = f2bf(p1);
    }
    // intra-wave LDS relayout fence (C-layout -> A-layout)
    asm volatile("s_wait_dscnt 0" ::: "memory");
    v16bf ap = ldfrag(&pw[ln * 32 + hi * 16]);

    // ---- O += P(16x32) * V(32x64): 4 column tiles from transposed V ----
    {
      v16bf bv;
      bv = ldfrag(vp + (size_t)(0 * 16 + ln) * S_ + kb + hi * 16); o0 = wmma_bf16(ap, bv, o0);
      bv = ldfrag(vp + (size_t)(1 * 16 + ln) * S_ + kb + hi * 16); o1 = wmma_bf16(ap, bv, o1);
      bv = ldfrag(vp + (size_t)(2 * 16 + ln) * S_ + kb + hi * 16); o2 = wmma_bf16(ap, bv, o2);
      bv = ldfrag(vp + (size_t)(3 * 16 + ln) * S_ + kb + hi * 16); o3 = wmma_bf16(ap, bv, o3);
    }
  }

  // ---- normalize + write bf16 [B,S,E] ----
#pragma unroll
  for (int r = 0; r < 8; ++r) {
    const int row = rowbase + r + hi * 8;
    const float inv = 1.0f / lstat[r];
    const size_t rb = ((size_t)b * S_ + row) * E_ + (size_t)h * D_;
    OUT[rb +  0 + ln] = f2bf(o0[r] * inv);
    OUT[rb + 16 + ln] = f2bf(o1[r] * inv);
    OUT[rb + 32 + ln] = f2bf(o2[r] * inv);
    OUT[rb + 48 + ln] = f2bf(o3[r] * inv);
  }
}

// ---------------------------------------------------------------------------
// Host-side orchestration
// ---------------------------------------------------------------------------
extern "C" void kernel_launch(void* const* d_in, const int* in_sizes, int n_in,
                              void* d_out, int out_size, void* d_ws, size_t ws_size,
                              hipStream_t stream) {
  const float* x  = (const float*)d_in[0];
  const unsigned char* mask = (const unsigned char*)d_in[1];
  const float* dt = (const float*)d_in[2];
  const float* wq = (const float*)d_in[3];
  const float* wk = (const float*)d_in[4];
  const float* wv = (const float*)d_in[5];
  const float* wo = (const float*)d_in[6];
  const float* bo = (const float*)d_in[7];
  float* out = (float*)d_out;

  // Workspace layout (bf16 elements). ATT aliases X (X dead after V GEMM).
  const size_t Mi = 1u << 20;
  unsigned short* W0 = (unsigned short*)d_ws;
  unsigned short* Xb  = W0;             // 8 Mi elems  (also ATT buffer later)
  unsigned short* WQb = W0 + 8  * Mi;   // 1 Mi each
  unsigned short* WKb = W0 + 9  * Mi;
  unsigned short* WVb = W0 + 10 * Mi;
  unsigned short* WOb = W0 + 11 * Mi;
  unsigned short* Qb  = W0 + 12 * Mi;   // 8 Mi
  unsigned short* Kb  = W0 + 20 * Mi;   // 8 Mi
  unsigned short* Vtb = W0 + 28 * Mi;   // 8 Mi  (total 36 Mi elems = 72 MB)
  unsigned short* Att = Xb;

  // 1) fp32 -> bf16
  cvt_bf16_k<<<(int)((8 * Mi / 4 + 255) / 256), 256, 0, stream>>>(x,  Xb,  (int)(8 * Mi / 4));
  cvt_bf16_k<<<(int)((Mi / 4 + 255) / 256),     256, 0, stream>>>(wq, WQb, (int)(Mi / 4));
  cvt_bf16_k<<<(int)((Mi / 4 + 255) / 256),     256, 0, stream>>>(wk, WKb, (int)(Mi / 4));
  cvt_bf16_k<<<(int)((Mi / 4 + 255) / 256),     256, 0, stream>>>(wv, WVb, (int)(Mi / 4));
  cvt_bf16_k<<<(int)((Mi / 4 + 255) / 256),     256, 0, stream>>>(wo, WOb, (int)(Mi / 4));

  // 2) QKV projections (+fused RoPE / V-transpose)
  dim3 gg(E_ / 64, M_ / 128);
  gemm_wmma_k<<<gg, 256, 0, stream>>>(Xb, WQb, Qb,  nullptr, dt, nullptr, 0);
  gemm_wmma_k<<<gg, 256, 0, stream>>>(Xb, WKb, Kb,  nullptr, dt, nullptr, 1);
  gemm_wmma_k<<<gg, 256, 0, stream>>>(Xb, WVb, Vtb, nullptr, dt, nullptr, 2);

  // 3) flash attention
  attn_wmma_k<<<dim3(S_ / 64, H_, B_), 128, 0, stream>>>(Qb, Kb, Vtb, mask, Att);

  // 4) output projection + bias -> fp32 d_out
  gemm_wmma_k<<<gg, 256, 0, stream>>>(Att, WOb, nullptr, out, dt, bo, 3);
}
